// GCN_87900800680758
// MI455X (gfx1250) — matrix operand order repaired and checked by reading, not compiled, not run on hardware
//
#include <hip/hip_runtime.h>
#include <hip/hip_bf16.h>

typedef __attribute__((ext_vector_type(16))) _Float16 v16h;
typedef __attribute__((ext_vector_type(8)))  _Float16 v8h;
typedef __attribute__((ext_vector_type(8)))  float    v8f;

#define HID 128
#define NHEAD 8
#define CH 16

// ---------------------------------------------------------------------------
// helpers
// ---------------------------------------------------------------------------
__device__ __forceinline__ void atomicMaxFloat(float* addr, float val) {
    // init value must be a negative float (-3e38). Standard ordered-int trick.
    if (val >= 0.f) atomicMax((int*)addr, __float_as_int(val));
    else            atomicMin((unsigned int*)addr, (unsigned int)__float_as_int(val));
}

// ---------------------------------------------------------------------------
// edge_attr column sums (for self-loop fill value = mean)
// ---------------------------------------------------------------------------
__global__ void attr_sum_kernel(const float* __restrict__ eattr, float* asum, int E) {
    __shared__ float s0[256], s1[256];
    int tid = threadIdx.x;
    long long i = (long long)blockIdx.x * 256 + tid;
    float a0 = 0.f, a1 = 0.f;
    if (i < E) { a0 = eattr[i * 2]; a1 = eattr[i * 2 + 1]; }
    s0[tid] = a0; s1[tid] = a1;
    __syncthreads();
    for (int st = 128; st > 0; st >>= 1) {
        if (tid < st) { s0[tid] += s0[tid + st]; s1[tid] += s1[tid + st]; }
        __syncthreads();
    }
    if (tid == 0) { atomicAdd(asum, s0[0]); atomicAdd(asum + 1, s1[0]); }
}

__global__ void convert_f16_kernel(const float* __restrict__ in, _Float16* __restrict__ out, size_t n) {
    size_t i = (size_t)blockIdx.x * blockDim.x + threadIdx.x;
    if (i < n) out[i] = (_Float16)in[i];
}

__global__ void fill_kernel(float* __restrict__ p, float v, size_t n) {
    size_t i = (size_t)blockIdx.x * blockDim.x + threadIdx.x;
    if (i < n) p[i] = v;
}

// ---------------------------------------------------------------------------
// Fused triple GEMM: xl = A@Wl+bl ; xr = A@Wr+br ; acc = A@Wk+bk+at_bias
// A is Npad x KDIM (f16), weights are KDIM x 128 (f32 in global, f16 in LDS).
// One wave computes a 16-row x 128-col stripe via v_wmma_f32_16x16x32_f16.
// All row buffers are padded to a multiple of 128 rows -> no store guards.
// ---------------------------------------------------------------------------
template <int KDIM>
__global__ __launch_bounds__(256) void gemm3_kernel(
    const _Float16* __restrict__ A,
    const float* __restrict__ Wl, const float* __restrict__ bl,
    const float* __restrict__ Wr, const float* __restrict__ br,
    const float* __restrict__ Wk, const float* __restrict__ bk,
    const float* __restrict__ at_bias,
    float* __restrict__ oxl, float* __restrict__ oxr, float* __restrict__ oacc)
{
    constexpr int KSTEPS = (KDIM + 31) / 32;
    constexpr int LDSTR  = 136;                 // halves; 16B-aligned frag bases
    __shared__ _Float16 sB[128 * LDSTR];        // transposed: sB[n*LDSTR + k]

    const int tid  = threadIdx.x;
    const int wave = tid >> 5;
    const int lane = tid & 31;
    const int m0   = blockIdx.x * 128 + wave * 16;

    // ---- A fragments (ISA 16-bit A 16x32 layout) ----
    // lane = row within tile; half-wave selects K sub-block (0 / 8).
    const int  mrow  = m0 + (lane & 15);
    const int  kbase = (lane < 16) ? 0 : 8;
    const _Float16* pA = A + (size_t)mrow * KDIM;

    v16h afrag[KSTEPS];
    #pragma unroll
    for (int s = 0; s < KSTEPS; ++s) {
        v8h lo = {}, hi = {};
        if ((s * 32 + 16) <= KDIM)              // folds at compile time
            lo = *(const v8h*)(pA + s * 32 + kbase);
        if ((s * 32 + 32) <= KDIM)
            hi = *(const v8h*)(pA + s * 32 + 16 + kbase);
        v16h f;
        #pragma unroll
        for (int j = 0; j < 8; ++j) { f[j] = lo[j]; f[8 + j] = hi[j]; }
        afrag[s] = f;
    }

    const float* Ws[3]   = { Wl, Wr, Wk };
    const float* bs[3]   = { bl, br, bk };
    float*       outs[3] = { oxl, oxr, oacc };

    const int ncol  = lane & 15;
    const int kbb   = (lane < 16) ? 0 : 8;
    const int rbase = m0 + ((lane < 16) ? 0 : 8);

    #pragma unroll
    for (int w = 0; w < 3; ++w) {
        __syncthreads();
        const float* W = Ws[w];
        for (int i = tid; i < 128 * 128; i += 256) {
            int k = i >> 7, n = i & 127;
            float v = (k < KDIM) ? W[k * 128 + n] : 0.f;
            sB[n * LDSTR + k] = (_Float16)v;
        }
        __syncthreads();

        float* __restrict__ op = outs[w];
        #pragma unroll
        for (int nt = 0; nt < 8; ++nt) {
            v8f c = {};
            #pragma unroll
            for (int s = 0; s < KSTEPS; ++s) {
                const _Float16* pb = &sB[(nt * 16 + ncol) * LDSTR + s * 32 + kbb];
                v8h blo = *(const v8h*)(pb);
                v8h bhi = *(const v8h*)(pb + 16);
                v16h bf;
                #pragma unroll
                for (int j = 0; j < 8; ++j) { bf[j] = blo[j]; bf[8 + j] = bhi[j]; }
                c = __builtin_amdgcn_wmma_f32_16x16x32_f16(
                        false, afrag[s], false, bf, (short)0, c, false, false);
            }
            const int col = nt * 16 + ncol;
            float bia = bs[w][col];
            if (w == 2) bia += at_bias[col];    // folds at compile time
            #pragma unroll
            for (int r = 0; r < 8; ++r)
                op[(size_t)(rbase + r) * 128 + col] = c[r] + bia;
        }
    }
}

// ---------------------------------------------------------------------------
// Edge kernels (GATv2 attention). Edges e<E from edge_index, e>=E self-loops.
// ---------------------------------------------------------------------------
__global__ void edge_logits_kernel(
    const float* __restrict__ xl, const float* __restrict__ xr,
    const int* __restrict__ eidx, const float* __restrict__ eattr,
    const float* __restrict__ asum,
    const float* __restrict__ We, const float* __restrict__ att,
    float* __restrict__ logits, float* __restrict__ mx, int E, int N)
{
    size_t tid = (size_t)blockIdx.x * blockDim.x + threadIdx.x;
    size_t total = (size_t)(E + N) * NHEAD;
    if (tid >= total) return;
    int h = (int)(tid & 7);
    size_t e = tid >> 3;
    int src, dst; float a0, a1;
    if (e < (size_t)E) {
        src = eidx[e]; dst = eidx[(size_t)E + e];
        a0 = eattr[e * 2]; a1 = eattr[e * 2 + 1];
    } else {
        src = dst = (int)(e - E);
        float inv = 1.f / (float)E;
        a0 = asum[0] * inv; a1 = asum[1] * inv;
    }
    const float* pl = xl + (size_t)src * HID + h * CH;
    const float* pr = xr + (size_t)dst * HID + h * CH;
    float lg = 0.f;
    #pragma unroll
    for (int c = 0; c < CH; ++c) {
        int hc = h * CH + c;
        float m = pl[c] + pr[c] + a0 * We[hc] + a1 * We[HID + hc];
        m = (m > 0.f) ? m : 0.2f * m;            // leaky_relu
        lg += m * att[hc];
    }
    logits[e * NHEAD + h] = lg;
    atomicMaxFloat(mx + (size_t)dst * NHEAD + h, lg);
}

__global__ void edge_softmax_kernel(
    float* __restrict__ logits, const float* __restrict__ mx,
    float* __restrict__ denom, const int* __restrict__ eidx, int E, int N)
{
    size_t tid = (size_t)blockIdx.x * blockDim.x + threadIdx.x;
    size_t total = (size_t)(E + N) * NHEAD;
    if (tid >= total) return;
    int h = (int)(tid & 7);
    size_t e = tid >> 3;
    int dst = (e < (size_t)E) ? eidx[(size_t)E + e] : (int)(e - E);
    float ea = expf(logits[tid] - mx[(size_t)dst * NHEAD + h]);
    logits[tid] = ea;                            // overwrite: now holds exp()
    atomicAdd(denom + (size_t)dst * NHEAD + h, ea);
}

__global__ void edge_aggregate_kernel(
    const float* __restrict__ xl, const float* __restrict__ ea,
    const float* __restrict__ denom, const int* __restrict__ eidx,
    float* __restrict__ acc, int E, int N)
{
    size_t tid = (size_t)blockIdx.x * blockDim.x + threadIdx.x;
    size_t total = (size_t)(E + N) * HID;
    if (tid >= total) return;
    int hc = (int)(tid & 127);
    size_t e = tid >> 7;
    int h = hc >> 4;
    int src, dst;
    if (e < (size_t)E) { src = eidx[e]; dst = eidx[(size_t)E + e]; }
    else               { src = dst = (int)(e - E); }
    float w = ea[e * NHEAD + h];
    float d = denom[(size_t)dst * NHEAD + h];
    float alpha = w / fmaxf(d, 1e-16f);
    atomicAdd(acc + (size_t)dst * HID + hc, xl[(size_t)src * HID + hc] * alpha);
}

__global__ void elu_kernel(float* __restrict__ p, size_t n) {
    size_t i = (size_t)blockIdx.x * blockDim.x + threadIdx.x;
    if (i < n) {
        float v = p[i];
        p[i] = (v > 0.f) ? v : (expf(v) - 1.f);
    }
}

// ---------------------------------------------------------------------------
// Global mean pool + readout
// ---------------------------------------------------------------------------
__global__ void pool_kernel(const float* __restrict__ h, const int* __restrict__ batch,
                            float* __restrict__ g, float* __restrict__ cnt, int N)
{
    size_t tid = (size_t)blockIdx.x * blockDim.x + threadIdx.x;
    size_t total = (size_t)N * HID;
    if (tid >= total) return;
    int hc = (int)(tid & 127);
    size_t n = tid >> 7;
    int b = batch[n];
    atomicAdd(g + (size_t)b * HID + hc, h[tid]);
    if (hc == 0) atomicAdd(cnt + b, 1.f);
}

__global__ void pool_div_kernel(float* __restrict__ g, const float* __restrict__ cnt, int G) {
    size_t tid = (size_t)blockIdx.x * blockDim.x + threadIdx.x;
    if (tid >= (size_t)G * HID) return;
    size_t gr = tid >> 7;
    g[tid] /= fmaxf(cnt[gr], 1.f);
}

__global__ void readout_kernel(const float* __restrict__ g, const float* __restrict__ W,
                               const float* __restrict__ b, float* __restrict__ out,
                               int G, int S)
{
    size_t tid = (size_t)blockIdx.x * blockDim.x + threadIdx.x;
    if (tid >= (size_t)G * S) return;
    int sp = (int)(tid % S);
    size_t gr = tid / S;
    const float* pg = g + gr * HID;
    float acc = b[sp];
    #pragma unroll 8
    for (int c = 0; c < HID; ++c) acc += pg[c] * W[(size_t)c * S + sp];
    out[tid] = acc;
}

// ---------------------------------------------------------------------------
// host driver
// ---------------------------------------------------------------------------
extern "C" void kernel_launch(void* const* d_in, const int* in_sizes, int n_in,
                              void* d_out, int out_size, void* d_ws, size_t ws_size,
                              hipStream_t stream) {
    const float* x       = (const float*)d_in[0];
    const int*   eidx    = (const int*)  d_in[1];
    const float* eattr   = (const float*)d_in[2];
    const int*   batch   = (const int*)  d_in[3];
    const float* at1_Wl  = (const float*)d_in[4];
    const float* at1_bl  = (const float*)d_in[5];
    const float* at1_Wr  = (const float*)d_in[6];
    const float* at1_br  = (const float*)d_in[7];
    const float* at1_We  = (const float*)d_in[8];
    const float* at1_att = (const float*)d_in[9];
    const float* at1_bias= (const float*)d_in[10];
    const float* l1_W    = (const float*)d_in[11];
    const float* l1_b    = (const float*)d_in[12];
    const float* atk_Wl  = (const float*)d_in[13];
    const float* atk_bl  = (const float*)d_in[14];
    const float* atk_Wr  = (const float*)d_in[15];
    const float* atk_br  = (const float*)d_in[16];
    const float* atk_We  = (const float*)d_in[17];
    const float* atk_att = (const float*)d_in[18];
    const float* atk_bias= (const float*)d_in[19];
    const float* lk_W    = (const float*)d_in[20];
    const float* lk_b    = (const float*)d_in[21];
    const float* lin_W   = (const float*)d_in[22];
    const float* lin_b   = (const float*)d_in[23];

    const int N    = in_sizes[0] / 16;
    const int Npad = (N + 127) & ~127;          // padded rows: no store guards
    const int E    = in_sizes[1] / 2;
    const int S    = in_sizes[23];
    const int G    = out_size / S;
    const int Et   = E + N;

    // workspace carve-out (row buffers padded to Npad)
    char* base = (char*)d_ws;
    size_t off = 0;
    auto take = [&](size_t bytes) -> char* {
        off = (off + 255) & ~(size_t)255;
        char* p = base + off;
        off += bytes;
        return p;
    };
    float*    hA     = (float*)   take((size_t)Npad * HID * 4);
    float*    hB     = (float*)   take((size_t)Npad * HID * 4);
    float*    xl     = (float*)   take((size_t)Npad * HID * 4);
    float*    xr     = (float*)   take((size_t)Npad * HID * 4);
    _Float16* hf16   = (_Float16*)take((size_t)Npad * HID * 2);
    float*    logits = (float*)   take((size_t)Et * NHEAD * 4);
    float*    mx     = (float*)   take((size_t)N * NHEAD * 4);
    float*    denom  = (float*)   take((size_t)N * NHEAD * 4);
    float*    g      = (float*)   take((size_t)G * HID * 4);
    float*    cnt    = (float*)   take((size_t)G * 4);
    float*    asum   = (float*)   take(2 * 4);
    (void)ws_size; (void)n_in;

    auto blocks = [](size_t n, int b) { return (unsigned)((n + b - 1) / b); };

    hipMemsetAsync(asum, 0, 8, stream);
    attr_sum_kernel<<<blocks(E, 256), 256, 0, stream>>>(eattr, asum, E);

    float* hcur = hA;
    float* acc  = hB;
    for (int layer = 0; layer < 10; ++layer) {
        const float *Wl, *bl, *Wr, *br, *We, *att, *atb, *Wk, *bk;
        int K;
        const float* Ain;
        if (layer == 0) {
            K = 16; Ain = x;
            Wl = at1_Wl; bl = at1_bl; Wr = at1_Wr; br = at1_br;
            We = at1_We; att = at1_att; atb = at1_bias;
            Wk = l1_W;   bk = l1_b;
        } else {
            K = HID; Ain = hcur;
            int i = layer - 1;
            Wl  = atk_Wl  + (size_t)i * HID * HID;
            bl  = atk_bl  + (size_t)i * HID;
            Wr  = atk_Wr  + (size_t)i * HID * HID;
            br  = atk_br  + (size_t)i * HID;
            We  = atk_We  + (size_t)i * 2 * HID;
            att = atk_att + (size_t)i * NHEAD * CH;
            atb = atk_bias+ (size_t)i * HID;
            Wk  = lk_W    + (size_t)i * HID * HID;
            bk  = lk_b    + (size_t)i * HID;
        }

        size_t nconv = (size_t)N * K;
        convert_f16_kernel<<<blocks(nconv, 256), 256, 0, stream>>>(Ain, hf16, nconv);

        unsigned ggrid = (unsigned)(Npad / 128);
        if (layer == 0)
            gemm3_kernel<16><<<ggrid, 256, 0, stream>>>(hf16, Wl, bl, Wr, br,
                                                        Wk, bk, atb, xl, xr, acc);
        else
            gemm3_kernel<128><<<ggrid, 256, 0, stream>>>(hf16, Wl, bl, Wr, br,
                                                         Wk, bk, atb, xl, xr, acc);

        fill_kernel<<<blocks((size_t)N * NHEAD, 256), 256, 0, stream>>>(mx, -3.0e38f, (size_t)N * NHEAD);
        hipMemsetAsync(denom, 0, (size_t)N * NHEAD * 4, stream);

        size_t tE8 = (size_t)Et * NHEAD;
        edge_logits_kernel<<<blocks(tE8, 256), 256, 0, stream>>>(
            xl, xr, eidx, eattr, asum, We, att, logits, mx, E, N);
        edge_softmax_kernel<<<blocks(tE8, 256), 256, 0, stream>>>(
            logits, mx, denom, eidx, E, N);

        size_t tE128 = (size_t)Et * HID;
        edge_aggregate_kernel<<<blocks(tE128, 256), 256, 0, stream>>>(
            xl, logits, denom, eidx, acc, E, N);

        elu_kernel<<<blocks((size_t)N * HID, 256), 256, 0, stream>>>(acc, (size_t)N * HID);

        float* t = hcur; hcur = acc; acc = t;
    }

    hipMemsetAsync(g, 0, (size_t)G * HID * 4, stream);
    hipMemsetAsync(cnt, 0, (size_t)G * 4, stream);
    pool_kernel<<<blocks((size_t)N * HID, 256), 256, 0, stream>>>(hcur, batch, g, cnt, N);
    pool_div_kernel<<<blocks((size_t)G * HID, 256), 256, 0, stream>>>(g, cnt, G);
    readout_kernel<<<blocks((size_t)G * S, 256), 256, 0, stream>>>(g, lin_W, lin_b, (float*)d_out, G, S);
}
